// TwoLayerGSCSNN_15831249453647
// MI455X (gfx1250) — 16-bit and 8-bit matrix operand order repaired and checked by reading, compiled, not run on hardware
//
#include <hip/hip_runtime.h>

typedef __attribute__((ext_vector_type(16))) _Float16 v16h;
typedef __attribute__((ext_vector_type(8)))  float    v8f;
typedef __attribute__((ext_vector_type(8)))  int      v8i;

#define TSTEPS 200
#define CIN    40
#define H1     512
#define H2     256
#define OUTN   12
#define NWG    16
#define THREADS 512
#define TH_F   1.0f

#define WREC_LDS_BYTES (32 * 8 * 32 * 32)
#define SPK1_LDS_BYTES (8 * 32 * 32)
#define SPK2_LDS_BYTES (4 * 32 * 32)

__device__ __forceinline__ unsigned char cvt_e4m3(float f) {
  unsigned u = __float_as_uint(f);
  unsigned char sign = (unsigned char)((u >> 24) & 0x80u);
  unsigned expf = (u >> 23) & 0xffu;
  if (expf == 0xffu) return (unsigned char)(sign | 0x7f);
  int exp = (int)expf - 127;
  unsigned man = u & 0x7fffffu;
  if (exp >= 9) return (unsigned char)(sign | 0x7e);
  int e8 = exp + 7;
  if (e8 >= 1) {
    unsigned keep = man >> 20;
    unsigned rem  = man & 0xfffffu;
    if (rem > 0x80000u || (rem == 0x80000u && (keep & 1u))) keep++;
    if (keep == 8u) { keep = 0u; e8++; }
    if (e8 > 15 || (e8 == 15 && keep == 7u)) return (unsigned char)(sign | 0x7e);
    return (unsigned char)(sign | (unsigned char)((e8 << 3) | keep));
  } else {
    int shift = 14 - exp;
    if (shift >= 32) return sign;
    unsigned m = man | 0x800000u;
    unsigned d   = m >> shift;
    unsigned rem = m & ((1u << shift) - 1u);
    unsigned half = 1u << (shift - 1);
    if (rem > half || (rem == half && (d & 1u))) d++;
    if (d >= 8u) return (unsigned char)(sign | 0x08);
    return (unsigned char)(sign | (unsigned char)d);
  }
}

__global__ void snn_pack_b16(const float* __restrict__ W, _Float16* __restrict__ dst,
                             int Korig, int Norig, int KT, int NT) {
  int idx = blockIdx.x * blockDim.x + threadIdx.x;
  int total = NT * KT * 32 * 16;
  if (idx >= total) return;
  int e    = idx & 15;
  int lane = (idx >> 4) & 31;
  int t    = idx >> 9;
  int kt = t % KT, nt = t / KT;
  int c = lane >> 4, nl = lane & 15;
  int k = kt * 32 + 16 * (e >> 3) + c * 8 + (e & 7);
  int n = nt * 16 + nl;
  float v = (k < Korig && n < Norig) ? W[n * Korig + k] : 0.0f;
  dst[idx] = (_Float16)v;
}

__global__ void snn_pack_b8(const float* __restrict__ W, unsigned char* __restrict__ dst,
                            int Korig, int Norig, int KT, int NT) {
  int idx = blockIdx.x * blockDim.x + threadIdx.x;
  int total = NT * KT * 32 * 32;
  if (idx >= total) return;
  int e    = idx & 31;
  int lane = (idx >> 5) & 31;
  int t    = idx >> 10;
  int kt = t % KT, nt = t / KT;
  int c = lane >> 4, nl = lane & 15;
  int k = kt * 64 + 16 * (e >> 3) + c * 8 + (e & 7);
  int n = nt * 16 + nl;
  float v = (k < Korig && n < Norig) ? W[n * Korig + k] : 0.0f;
  dst[idx] = cvt_e4m3(v);
}

__global__ void snn_pack_x(const float* __restrict__ x, _Float16* __restrict__ dst) {
  int idx = blockIdx.x * blockDim.x + threadIdx.x;
  int total = TSTEPS * NWG * 2 * 32 * 16;
  if (idx >= total) return;
  int e    = idx & 15;
  int lane = (idx >> 4) & 31;
  int r    = idx >> 9;
  int kt = r & 1;  r >>= 1;
  int wg = r & 15; r >>= 4;
  int t  = r;
  int c = lane >> 4, m = lane & 15;
  int k = kt * 32 + 16 * (e >> 3) + c * 8 + (e & 7);
  int b = wg * 16 + m;
  float v = (k < CIN) ? x[(b * TSTEPS + t) * CIN + k] : 0.0f;
  dst[idx] = (_Float16)v;
}

__global__ __launch_bounds__(THREADS) void snn_scan(
    const _Float16*      __restrict__ xpack,
    const _Float16*      __restrict__ w1p,
    const unsigned char* __restrict__ wrec8,
    const unsigned char* __restrict__ w2p8,
    const unsigned char* __restrict__ wout8,
    const float* __restrict__ alpha1, const float* __restrict__ rho1, const float* __restrict__ ba1,
    const float* __restrict__ alpha2, const float* __restrict__ rho2, const float* __restrict__ ba2,
    const float* __restrict__ beta_out_p,
    float* __restrict__ out)
{
  __shared__ __align__(32) unsigned char smem[WREC_LDS_BYTES + SPK1_LDS_BYTES + SPK2_LDS_BYTES];
  unsigned char* wrec_s = smem;
  unsigned char* spk1_s = smem + WREC_LDS_BYTES;
  unsigned char* spk2_s = smem + WREC_LDS_BYTES + SPK1_LDS_BYTES;

  const int wg   = blockIdx.x;
  const int lane = threadIdx.x & 31;
  const int wave = threadIdx.x >> 5;
  const int nl   = lane & 15;
  const int cl   = lane >> 4;

  for (int i = threadIdx.x * 16; i < WREC_LDS_BYTES; i += THREADS * 16)
    *(int4*)(wrec_s + i) = *(const int4*)(wrec8 + i);
  for (int i = threadIdx.x * 4; i < SPK1_LDS_BYTES + SPK2_LDS_BYTES; i += THREADS * 4)
    *(int*)(spk1_s + i) = 0;

  float al1[2], rh1[2], bb1[2];
#pragma unroll
  for (int i = 0; i < 2; ++i) {
    int n = (wave * 2 + i) * 16 + nl;
    al1[i] = alpha1[n]; rh1[i] = rho1[n]; bb1[i] = ba1[n];
  }
  const int n2 = wave * 16 + nl;
  const float al2 = alpha2[n2], rh2 = rho2[n2], bb2 = ba2[n2];
  const float beta = beta_out_p[0];

  v8f v1[2], a1[2], v2, a2, v_out, out_sum;
#pragma unroll
  for (int i = 0; i < 2; ++i) { v1[i] = {}; a1[i] = {}; }
  v2 = {}; a2 = {}; v_out = {}; out_sum = {};

  __syncthreads();

  const v16h* xv  = (const v16h*)xpack;
  const v16h* w1v = (const v16h*)w1p;

  for (int t = 0; t < TSTEPS; ++t) {
    v8f acc[2];
#pragma unroll
    for (int i = 0; i < 2; ++i) acc[i] = {};

#pragma unroll
    for (int kt = 0; kt < 8; ++kt) {
      v8i a = *(const v8i*)(spk1_s + (kt * 32 + lane) * 32);
#pragma unroll
      for (int i = 0; i < 2; ++i) {
        int nt = wave * 2 + i;
        v8i b = *(const v8i*)(wrec_s + (((nt * 8 + kt) * 32) + lane) * 32);
        acc[i] = __builtin_amdgcn_wmma_f32_16x16x64_fp8_fp8(a, b, (short)0, acc[i],
                                                            false, false);
      }
    }
#pragma unroll
    for (int kt = 0; kt < 2; ++kt) {
      v16h a = xv[((t * NWG + wg) * 2 + kt) * 32 + lane];
#pragma unroll
      for (int i = 0; i < 2; ++i) {
        int nt = wave * 2 + i;
        v16h b = w1v[(nt * 2 + kt) * 32 + lane];
        acc[i] = __builtin_amdgcn_wmma_f32_16x16x32_f16(false, a, false, b,
                                                        (short)0, acc[i], false, false);
      }
    }
    __syncthreads();

#pragma unroll
    for (int i = 0; i < 2; ++i) {
      int k  = (wave * 2 + i) * 16 + nl;
      int kt = k >> 6;
      int kk = k & 63;
      int c2 = (kk & 15) >> 3;
      int e  = ((kk >> 4) << 3) + (kk & 7);
#pragma unroll
      for (int j = 0; j < 8; ++j) {
        float vv = v1[i][j];
        float sp = (vv >= TH_F) ? 1.0f : 0.0f;
        float vn = al1[i] * (vv - sp * TH_F) + (1.0f - al1[i]) * (acc[i][j] - a1[i][j]);
        float sn = (vn >= TH_F) ? 1.0f : 0.0f;
        a1[i][j] = rh1[i] * a1[i][j] + bb1[i] * sn;
        v1[i][j] = vn;
        int m = j + 8 * cl;
        spk1_s[kt * 1024 + (c2 * 16 + m) * 32 + e] =
            (vn >= TH_F) ? (unsigned char)0x38 : (unsigned char)0x00;
      }
    }
    __syncthreads();

    v8f acc2 = {};
#pragma unroll
    for (int kt = 0; kt < 8; ++kt) {
      v8i a = *(const v8i*)(spk1_s + (kt * 32 + lane) * 32);
      if (kt < 7)
        __builtin_prefetch((const void*)(w2p8 + (((wave * 8 + kt + 1) * 32) + lane) * 32), 0, 3);
      v8i b = *(const v8i*)(w2p8 + (((wave * 8 + kt) * 32) + lane) * 32);
      acc2 = __builtin_amdgcn_wmma_f32_16x16x64_fp8_fp8(a, b, (short)0, acc2, false, false);
    }

    {
      int k  = wave * 16 + nl;
      int kt = k >> 6;
      int kk = k & 63;
      int c2 = (kk & 15) >> 3;
      int e  = ((kk >> 4) << 3) + (kk & 7);
#pragma unroll
      for (int j = 0; j < 8; ++j) {
        float vn = al2 * v2[j] + (1.0f - al2) * (acc2[j] - a2[j]);
        float sn = (vn >= TH_F) ? 1.0f : 0.0f;
        a2[j] = rh2 * a2[j] + bb2 * sn;
        v2[j] = vn;
        int m = j + 8 * cl;
        spk2_s[kt * 1024 + (c2 * 16 + m) * 32 + e] =
            (vn >= TH_F) ? (unsigned char)0x38 : (unsigned char)0x00;
      }
    }
    __syncthreads();

    if (wave == 0) {
      v8f acc3 = {};
#pragma unroll
      for (int kt = 0; kt < 4; ++kt) {
        v8i a = *(const v8i*)(spk2_s + (kt * 32 + lane) * 32);
        v8i b = *(const v8i*)(wout8 + ((kt * 32) + lane) * 32);
        acc3 = __builtin_amdgcn_wmma_f32_16x16x64_fp8_fp8(a, b, (short)0, acc3, false, false);
      }
#pragma unroll
      for (int j = 0; j < 8; ++j) {
        v_out[j]   = beta * v_out[j] + (1.0f - beta) * acc3[j];
        out_sum[j] = out_sum[j] + v_out[j];
      }
    }
  }

  if (wave == 0) {
#pragma unroll
    for (int j = 0; j < 8; ++j) {
      int m = j + 8 * cl;
      if (nl < OUTN)
        out[(wg * 16 + m) * OUTN + nl] = out_sum[j] * (1.0f / (float)TSTEPS);
    }
  }
}

extern "C" void kernel_launch(void* const* d_in, const int* in_sizes, int n_in,
                              void* d_out, int out_size, void* d_ws, size_t ws_size,
                              hipStream_t stream) {
  const float* x      = (const float*)d_in[0];
  const float* W1     = (const float*)d_in[1];
  const float* Wrec   = (const float*)d_in[2];
  const float* W2     = (const float*)d_in[3];
  const float* Wout   = (const float*)d_in[4];
  const float* alpha1 = (const float*)d_in[5];
  const float* rho1   = (const float*)d_in[6];
  const float* ba1    = (const float*)d_in[7];
  const float* alpha2 = (const float*)d_in[8];
  const float* rho2   = (const float*)d_in[9];
  const float* ba2    = (const float*)d_in[10];
  const float* beta   = (const float*)d_in[11];
  float* out = (float*)d_out;

  char* ws = (char*)d_ws;
  size_t off = 0;
  auto carveB = [&](size_t bytes) -> unsigned char* {
    unsigned char* p = (unsigned char*)(ws + off);
    off += ((bytes + 255) / 256) * 256;
    return p;
  };
  _Float16*      xpack = (_Float16*)carveB((size_t)TSTEPS * NWG * 2 * 32 * 16 * 2);
  _Float16*      w1p   = (_Float16*)carveB((size_t)32 * 2 * 32 * 16 * 2);
  unsigned char* wrec8 = carveB((size_t)32 * 8 * 32 * 32);
  unsigned char* w2p8  = carveB((size_t)16 * 8 * 32 * 32);
  unsigned char* wout8 = carveB((size_t)1 * 4 * 32 * 32);

  {
    int total = TSTEPS * NWG * 2 * 512;
    snn_pack_x<<<(total + 255) / 256, 256, 0, stream>>>(x, xpack);
  }
  {
    int total = 32 * 2 * 512;
    snn_pack_b16<<<(total + 255) / 256, 256, 0, stream>>>(W1, w1p, CIN, H1, 2, 32);
  }
  {
    int total = 32 * 8 * 1024;
    snn_pack_b8<<<(total + 255) / 256, 256, 0, stream>>>(Wrec, wrec8, H1, H1, 8, 32);
  }
  {
    int total = 16 * 8 * 1024;
    snn_pack_b8<<<(total + 255) / 256, 256, 0, stream>>>(W2, w2p8, H1, H2, 8, 16);
  }
  {
    int total = 1 * 4 * 1024;
    snn_pack_b8<<<(total + 255) / 256, 256, 0, stream>>>(Wout, wout8, H2, OUTN, 4, 1);
  }

  snn_scan<<<NWG, THREADS, 0, stream>>>(xpack, w1p, wrec8, w2p8, wout8,
                                        alpha1, rho1, ba1, alpha2, rho2, ba2,
                                        beta, out);
}
